// MyTransformerEncoderLayer_23553600651348
// MI455X (gfx1250) — matrix operand order repaired and checked by reading, not compiled, not run on hardware
//
#include <hip/hip_runtime.h>

// ---------------------------------------------------------------------------
// MI455X (gfx1250) transformer encoder layer, bf16 WMMA everywhere.
// S=1024, N=8, E=1024, H=16, D=64, DFF=4096.  M = S*N = 8192 token rows.
// All GEMMs are C[M,N] = A[M,K] @ B[N,K]^T  (K contiguous in both operands).
// Weight (B) tiles are staged into LDS by the Tensor Data Mover with
// hardware padding; activation (A) tiles use vectorized global_load_b128.
// ---------------------------------------------------------------------------

typedef __attribute__((ext_vector_type(16))) __bf16 bf16x16;
typedef __attribute__((ext_vector_type(8)))  float  f32x8;
typedef __attribute__((ext_vector_type(4)))  unsigned int u32x4;
typedef __attribute__((ext_vector_type(8)))  int i32x8;
typedef __attribute__((ext_vector_type(4)))  int i32x4;

#define ZERO8 {0.f,0.f,0.f,0.f,0.f,0.f,0.f,0.f}

__device__ __forceinline__ f32x8 wmma_bf16(bf16x16 a, bf16x16 b, f32x8 c) {
  // v_wmma_f32_16x16x32_bf16  (neg_a, A, neg_b, B, c_mod, C, reuse_a, reuse_b)
  return __builtin_amdgcn_wmma_f32_16x16x32_bf16(false, a, false, b,
                                                 (short)0, c, false, false);
}

// ---------------------------------------------------------------------------
// TDM: 2D tile load Global -> LDS (CDNA5 ISA).
// D# group0: count=1 | lds_addr | global_addr | type=2.
// D# group1: data_size=2B, pad_enable (pad_amount DWORDs every pad_interval),
//            tensor_dim0/1, tile_dim0/1, tensor_dim0_stride.
// Tracked by TENSORcnt; caller waits with s_wait_tensorcnt then barriers.
// ---------------------------------------------------------------------------
#if __has_builtin(__builtin_amdgcn_tensor_load_to_lds)
#define USE_TDM 1
__device__ __forceinline__ void tdm_load_tile_2d(
    const void* gptr, unsigned lds_addr, unsigned tensor_d0, unsigned tensor_d1,
    unsigned tile_d0, unsigned tile_d1, unsigned long long stride0,
    unsigned pad_interval, unsigned pad_amount) {
  const unsigned long long ga = (unsigned long long)gptr;
  u32x4 g0;
  g0[0] = 1u;                                   // count=1, user descriptor
  g0[1] = lds_addr;                             // LDS byte address
  g0[2] = (unsigned)(ga & 0xFFFFFFFFu);         // global_addr[31:0]
  g0[3] = (unsigned)((ga >> 32) & 0x1FFFFFFu)   // global_addr[56:32]
        | (2u << 30);                           // type=2 ("image")
  i32x8 g1;
  g1[0] = (int)((1u << 16)                      // data_size=1 -> 2 bytes
        | (1u << 20)                            // pad_enable
        | (pad_interval << 22) | (pad_amount << 25));
  g1[1] = (int)((tensor_d0 & 0xFFFFu) << 16);   // atomic_barrier_addr=0
  g1[2] = (int)((tensor_d0 >> 16) | ((tensor_d1 & 0xFFFFu) << 16));
  g1[3] = (int)((tensor_d1 >> 16) | (tile_d0 << 16));
  g1[4] = (int)(tile_d1 & 0xFFFFu);             // tile_dim2 = 0
  g1[5] = (int)(stride0 & 0xFFFFFFFFull);       // tensor_dim0_stride[31:0]
  g1[6] = (int)((stride0 >> 32) & 0xFFFFull);   // [47:32]; dim1_stride lo16=0
  g1[7] = 0;
  const i32x4 z4 = {0, 0, 0, 0};
#if __clang_major__ >= 23
  const i32x8 z8 = {0, 0, 0, 0, 0, 0, 0, 0};
  __builtin_amdgcn_tensor_load_to_lds(g0, g1, z4, z4, z8, 0);
#else
  __builtin_amdgcn_tensor_load_to_lds(g0, g1, z4, z4, 0);
#endif
}
#else
#define USE_TDM 0
#endif

// A-matrix 16x32 bf16 fragment: lanes 0-15 row M=lane, K in
// {0..15}; lanes 16-31 row M=lane-16, K in {16..31}.
// Two contiguous 16B loads per lane.
__device__ __forceinline__ bf16x16 load_frag_a(const __bf16* base, int stride,
                                               int kb) {
  const int lane = threadIdx.x & 31;
  const int hf = lane >> 4;
  const int r = lane & 15;
  const __bf16* p = base + r * stride + kb + hf * 16;
  union { bf16x16 v; uint4 u[2]; } f;
  f.u[0] = *(const uint4*)(p);
  f.u[1] = *(const uint4*)(p + 8);
  return f.v;
}

// B-matrix 32x16 bf16 fragment, column-major load: lane holds one output
// column (= one row of the NT weight matrix), K contiguous (16 per half-wave).
__device__ __forceinline__ bf16x16 load_frag_b(const __bf16* base, int stride,
                                               int kb) {
  const int lane = threadIdx.x & 31;
  const int hf = lane >> 4;
  const int n = lane & 15;
  const __bf16* p = base + n * stride + kb + hf * 16;
  union { bf16x16 v; uint4 u[2]; } f;
  f.u[0] = *(const uint4*)(p);
  f.u[1] = *(const uint4*)(p + 8);
  return f.v;
}

__device__ __forceinline__ float wave_max(float v) {
  #pragma unroll
  for (int o = 16; o > 0; o >>= 1) v = fmaxf(v, __shfl_xor(v, o, 32));
  return v;
}
__device__ __forceinline__ float wave_sum(float v) {
  #pragma unroll
  for (int o = 16; o > 0; o >>= 1) v += __shfl_xor(v, o, 32);
  return v;
}

// ---------------------------------------------------------------------------
// f32 -> bf16 conversion (grid-stride)
// ---------------------------------------------------------------------------
__global__ void cvt_f32_bf16(const float* __restrict__ in,
                             __bf16* __restrict__ out, int n) {
  for (int i = blockIdx.x * blockDim.x + threadIdx.x; i < n;
       i += gridDim.x * blockDim.x)
    out[i] = (__bf16)in[i];
}

// ---------------------------------------------------------------------------
// Tiled NT GEMM: 128x128 tile, BK=32, 256 threads (8 waves), each wave a
// 64x32 sub-block = 4x2 WMMA accumulators.  B tile staged by TDM (wave 0),
// A tile by vectorized global loads.  LDS rows padded +8 halves (16 B) per
// 32-half row; TDM reproduces this with pad_interval=3 (64 B), pad_amount=3
// (4 DWORDs).
// ---------------------------------------------------------------------------
template <bool RELU, bool BF16_OUT, bool F32_OUT>
__global__ __launch_bounds__(256) void gemm_bf16_nt(
    const __bf16* __restrict__ A, const __bf16* __restrict__ B,
    const float* __restrict__ bias, float* __restrict__ Cf,
    __bf16* __restrict__ Cb, int M, int N, int K) {
  constexpr int BM = 128, BN = 128, BK = 32, LDT = BK + 8;  // pad: 16B aligned
  __shared__ __align__(16) __bf16 As[BM * LDT];
  __shared__ __align__(16) __bf16 Bs[BN * LDT];

  const int tid = threadIdx.x;
  const int mblk = blockIdx.y * BM;
  const int nblk = blockIdx.x * BN;
  const int w = tid >> 5, lane = tid & 31;
  const int wm = (w & 1) * 64;
  const int wn = (w >> 1) * 32;

  f32x8 acc[4][2];
  #pragma unroll
  for (int i = 0; i < 4; ++i)
    #pragma unroll
    for (int j = 0; j < 2; ++j) acc[i][j] = (f32x8)ZERO8;

  const int lrow = tid >> 1;          // 0..127
  const int lseg = (tid & 1) * 16;    // 0 / 16
#if USE_TDM
  const unsigned lds_b = (unsigned)(unsigned long long)(&Bs[0]);
#endif

  for (int kt = 0; kt < K; kt += BK) {
    // A tile: 256 threads x 32 B, vectorized global -> LDS
    const __bf16* ga = A + (size_t)(mblk + lrow) * K + kt + lseg;
    *(uint4*)&As[lrow * LDT + lseg]     = *(const uint4*)(ga);
    *(uint4*)&As[lrow * LDT + lseg + 8] = *(const uint4*)(ga + 8);
    if (kt + BK < K) __builtin_prefetch(ga + BK, 0, 0);  // global_prefetch_b8
#if USE_TDM
    // B tile: one TDM descriptor per K-step, issued by wave 0 only.
    if (tid < 32) {
      tdm_load_tile_2d(B + (size_t)nblk * K + kt, lds_b,
                       (unsigned)K, (unsigned)BN,   // tensor dims (elements)
                       BK, BN,                      // tile 32 x 128
                       (unsigned long long)K,       // row stride (elements)
                       /*pad_interval=*/3,          // every 16 DWORDs (64 B)
                       /*pad_amount=*/3);           // +4 DWORDs (16 B)
      __builtin_amdgcn_s_wait_tensorcnt(0);
    }
#else
    const __bf16* gb = B + (size_t)(nblk + lrow) * K + kt + lseg;
    *(uint4*)&Bs[lrow * LDT + lseg]     = *(const uint4*)(gb);
    *(uint4*)&Bs[lrow * LDT + lseg + 8] = *(const uint4*)(gb + 8);
#endif
    __syncthreads();

    bf16x16 af[4], bfv[2];
    #pragma unroll
    for (int mi = 0; mi < 4; ++mi)
      af[mi] = load_frag_a(&As[(wm + mi * 16) * LDT], LDT, 0);
    #pragma unroll
    for (int ni = 0; ni < 2; ++ni)
      bfv[ni] = load_frag_b(&Bs[(wn + ni * 16) * LDT], LDT, 0);
    #pragma unroll
    for (int mi = 0; mi < 4; ++mi)
      #pragma unroll
      for (int ni = 0; ni < 2; ++ni)
        acc[mi][ni] = wmma_bf16(af[mi], bfv[ni], acc[mi][ni]);
    __syncthreads();
  }

  const int hf = lane >> 4, nn = lane & 15;
  #pragma unroll
  for (int mi = 0; mi < 4; ++mi) {
    #pragma unroll
    for (int ni = 0; ni < 2; ++ni) {
      const int col = nblk + wn + ni * 16 + nn;
      const float bv = bias ? bias[col] : 0.f;
      #pragma unroll
      for (int i = 0; i < 8; ++i) {
        const int row = mblk + wm + mi * 16 + hf * 8 + i;
        float v = acc[mi][ni][i] + bv;
        if (RELU) v = fmaxf(v, 0.f);
        if (F32_OUT) Cf[(size_t)row * N + col] = v;
        if (BF16_OUT) Cb[(size_t)row * N + col] = (__bf16)v;
      }
    }
  }
}

// ---------------------------------------------------------------------------
// Fused attention: one workgroup owns (batch n, 16-row block of S) and loops
// all 16 heads.  Scores strip (16x1024 f32) lives in LDS, softmax in place,
// head-average accumulated straight into d_out (exclusive rows -> no atomics),
// then AV via WMMA.  128 threads = 4 waves; each wave owns 16 of D=64 dims.
// qkv layout: [M=8192, 3E]: q at h*64, k at E + h*64, v at 2E + h*64.
// ---------------------------------------------------------------------------
__global__ __launch_bounds__(128) void attn_fused(
    const __bf16* __restrict__ qkv, __bf16* __restrict__ ctx,
    float* __restrict__ attn_avg) {
  constexpr int S = 1024, NB = 8, E = 1024, H = 16, D = 64, TE = 3072;
  constexpr int QST = 72, PST = 1032, VST = 72;  // padded strides (16B mult)
  __shared__ __align__(16) __bf16 Qs[16 * QST];
  __shared__ __align__(16) float  Ps[16 * PST];
  __shared__ __align__(16) __bf16 Vs[32 * VST];

  const int sb = blockIdx.x;   // 0..63  (16-row block)
  const int n  = blockIdx.y;   // 0..7
  const int tid = threadIdx.x;
  const int w = tid >> 5, lane = tid & 31;
  const int hf = lane >> 4, ln = lane & 15;
  const float scale = 0.125f;  // 1/sqrt(64)

  for (int h = 0; h < H; ++h) {
    // ---- load Q block 16x64 -> LDS
    {
      const int row = tid >> 3;         // 0..15
      const int seg = (tid & 7) * 8;    // 0..56
      const int s = sb * 16 + row;
      const __bf16* gq = qkv + (size_t)(s * NB + n) * TE + h * D + seg;
      *(uint4*)&Qs[row * QST + seg] = *(const uint4*)gq;
    }
    __syncthreads();

    // ---- scores = scale * Q K^T  (16 x 1024), WMMA over D=64 (2 steps)
    for (int tt = w; tt < 64; tt += 4) {
      const int tbase = tt * 16;
      f32x8 acc = (f32x8)ZERO8;
      #pragma unroll
      for (int kb = 0; kb < D; kb += 32) {
        bf16x16 a = load_frag_a(Qs, QST, kb);
        const int t = tbase + ln;       // this lane's K-row (output column)
        const __bf16* gk =
            qkv + (size_t)(t * NB + n) * TE + E + h * D + kb + hf * 16;
        union { bf16x16 v; uint4 u[2]; } bfv;
        bfv.u[0] = *(const uint4*)gk;
        bfv.u[1] = *(const uint4*)(gk + 8);
        acc = wmma_bf16(a, bfv.v, acc);
      }
      #pragma unroll
      for (int i = 0; i < 8; ++i)
        Ps[(hf * 8 + i) * PST + tbase + ln] = acc[i] * scale;
    }
    __syncthreads();

    // ---- softmax (wave per row, 4 rows each) + head-average to global
    for (int j = 0; j < 4; ++j) {
      const int r = w * 4 + j;
      float* Pr = &Ps[r * PST];
      float m = -3.0e38f;
      for (int c = lane; c < S; c += 32) m = fmaxf(m, Pr[c]);
      m = wave_max(m);
      float sum = 0.f;
      for (int c = lane; c < S; c += 32) {
        const float e0 = __expf(Pr[c] - m);
        Pr[c] = e0;
        sum += e0;
      }
      sum = wave_sum(sum);
      const float inv = 1.f / sum;
      const int s = sb * 16 + r;
      float* avg = attn_avg + ((size_t)n * S + s) * S;
      for (int c = lane; c < S; c += 32) {
        const float p = Pr[c] * inv;
        Pr[c] = p;
        float a0 = p * (1.0f / 16.0f);
        if (h) a0 += avg[c];   // rows exclusive to this WG: plain RMW
        avg[c] = a0;
      }
    }
    __syncthreads();

    // ---- ctx = P @ V  (16 x 64); wave owns 16 dims, K loop over S in 32s
    f32x8 acc = (f32x8)ZERO8;
    const int dbase = w * 16;
    for (int tt = 0; tt < 32; ++tt) {
      const int tb = tt * 32;
      {  // cooperative V tile 32x64 -> LDS
        const int row = tid >> 2;        // 0..31
        const int seg = (tid & 3) * 16;  // 0..48
        const __bf16* gv =
            qkv + (size_t)((tb + row) * NB + n) * TE + 2 * E + h * D + seg;
        *(uint4*)&Vs[row * VST + seg]     = *(const uint4*)gv;
        *(uint4*)&Vs[row * VST + seg + 8] = *(const uint4*)(gv + 8);
      }
      __syncthreads();
      union { bf16x16 v; __bf16 e[16]; } af;
      #pragma unroll
      for (int q = 0; q < 16; ++q) {
        const int k = hf * 16 + q;
        af.e[q] = (__bf16)Ps[ln * PST + tb + k];
      }
      union { bf16x16 v; __bf16 e[16]; } bfv;
      #pragma unroll
      for (int q = 0; q < 16; ++q)
        bfv.e[q] = Vs[(hf * 16 + q) * VST + dbase + ln];
      acc = wmma_bf16(af.v, bfv.v, acc);
      __syncthreads();
    }
    #pragma unroll
    for (int i = 0; i < 8; ++i) {
      const int m = hf * 8 + i;
      const int s = sb * 16 + m;
      ctx[(size_t)(s * NB + n) * E + h * D + dbase + ln] = (__bf16)acc[i];
    }
    __syncthreads();
  }
}

// ---------------------------------------------------------------------------
// Fused residual add + LayerNorm; optional bf16 mirror for the next GEMM.
// One block per row (E=1024), 256 threads x 4 elems.
// ---------------------------------------------------------------------------
__global__ __launch_bounds__(256) void add_ln(
    const float* __restrict__ a, const float* __restrict__ b,
    const float* __restrict__ g, const float* __restrict__ beta,
    float* __restrict__ outf, __bf16* __restrict__ outb) {
  constexpr int E = 1024;
  constexpr float EPS = 1e-5f;
  __shared__ float rs[8];
  __shared__ float rq[8];
  const size_t row = blockIdx.x;
  const float* ar = a + row * E;
  const float* br = b + row * E;
  float v[4], sum = 0.f, sq = 0.f;
  #pragma unroll
  for (int k = 0; k < 4; ++k) {
    const int c = threadIdx.x + k * 256;
    const float t = ar[c] + br[c];
    v[k] = t;
    sum += t;
    sq += t * t;
  }
  sum = wave_sum(sum);
  sq = wave_sum(sq);
  const int w = threadIdx.x >> 5, lane = threadIdx.x & 31;
  if (lane == 0) { rs[w] = sum; rq[w] = sq; }
  __syncthreads();
  if (w == 0) {
    float s2 = lane < 8 ? rs[lane] : 0.f;
    float q2 = lane < 8 ? rq[lane] : 0.f;
    s2 = wave_sum(s2);
    q2 = wave_sum(q2);
    if (lane == 0) { rs[0] = s2; rq[0] = q2; }
  }
  __syncthreads();
  const float mean = rs[0] * (1.f / E);
  const float var = rq[0] * (1.f / E) - mean * mean;
  const float rstd = rsqrtf(var + EPS);
  #pragma unroll
  for (int k = 0; k < 4; ++k) {
    const int c = threadIdx.x + k * 256;
    const float y = (v[k] - mean) * rstd * g[c] + beta[c];
    outf[row * E + c] = y;
    if (outb) outb[row * E + c] = (__bf16)y;
  }
}

// ---------------------------------------------------------------------------
extern "C" void kernel_launch(void* const* d_in, const int* in_sizes, int n_in,
                              void* d_out, int out_size, void* d_ws,
                              size_t ws_size, hipStream_t stream) {
  (void)in_sizes; (void)n_in; (void)out_size; (void)ws_size;
  constexpr int S = 1024, NB = 8, E = 1024, DFF = 4096;
  constexpr int M = S * NB;  // 8192

  const float* src       = (const float*)d_in[0];
  const float* in_proj_w = (const float*)d_in[1];
  const float* in_proj_b = (const float*)d_in[2];
  const float* out_w     = (const float*)d_in[3];
  const float* out_b     = (const float*)d_in[4];
  const float* w1        = (const float*)d_in[5];
  const float* b1        = (const float*)d_in[6];
  const float* w2        = (const float*)d_in[7];
  const float* b2        = (const float*)d_in[8];
  const float* ln1_g     = (const float*)d_in[9];
  const float* ln1_b     = (const float*)d_in[10];
  const float* ln2_g     = (const float*)d_in[11];
  const float* ln2_b     = (const float*)d_in[12];

  float* x_out = (float*)d_out;                          // [S,N,E]
  float* attn_avg = (float*)d_out + (size_t)M * E;       // [N,S,S]

  char* p = (char*)d_ws;
  auto take = [&](size_t bytes) -> void* {
    void* r = (void*)p;
    p += (bytes + 255) & ~(size_t)255;
    return r;
  };
  __bf16* src_b  = (__bf16*)take((size_t)M * E * 2);
  __bf16* wqkv_b = (__bf16*)take((size_t)3 * E * E * 2);
  __bf16* outw_b = (__bf16*)take((size_t)E * E * 2);
  __bf16* w1_b   = (__bf16*)take((size_t)DFF * E * 2);
  __bf16* w2_b   = (__bf16*)take((size_t)E * DFF * 2);
  __bf16* qkv_b  = (__bf16*)take((size_t)M * 3 * E * 2);
  __bf16* ctx_b  = (__bf16*)take((size_t)M * E * 2);
  float*  ao_f   = (float*)take((size_t)M * E * 4);
  float*  x1_f   = (float*)take((size_t)M * E * 4);
  __bf16* x1_b   = (__bf16*)take((size_t)M * E * 2);
  __bf16* h1_b   = (__bf16*)take((size_t)M * DFF * 2);
  float*  ff_f   = (float*)take((size_t)M * E * 4);

  // f32 -> bf16 staging
  cvt_f32_bf16<<<2048, 256, 0, stream>>>(src, src_b, M * E);
  cvt_f32_bf16<<<2048, 256, 0, stream>>>(in_proj_w, wqkv_b, 3 * E * E);
  cvt_f32_bf16<<<1024, 256, 0, stream>>>(out_w, outw_b, E * E);
  cvt_f32_bf16<<<2048, 256, 0, stream>>>(w1, w1_b, DFF * E);
  cvt_f32_bf16<<<2048, 256, 0, stream>>>(w2, w2_b, E * DFF);

  // qkv = src @ in_proj_w^T + b      [8192, 3072]
  gemm_bf16_nt<false, true, false><<<dim3(3 * E / 128, M / 128), 256, 0,
                                     stream>>>(src_b, wqkv_b, in_proj_b,
                                               nullptr, qkv_b, M, 3 * E, E);

  // fused attention -> ctx bf16, head-avg attention -> d_out tail
  attn_fused<<<dim3(S / 16, NB), 128, 0, stream>>>(qkv_b, ctx_b, attn_avg);

  // attn_out = ctx @ out_w^T + b     [8192, 1024] f32
  gemm_bf16_nt<false, false, true><<<dim3(E / 128, M / 128), 256, 0, stream>>>(
      ctx_b, outw_b, out_b, ao_f, nullptr, M, E, E);

  // x1 = LN(src + attn_out)
  add_ln<<<M, 256, 0, stream>>>(src, ao_f, ln1_g, ln1_b, x1_f, x1_b);

  // h1 = relu(x1 @ w1^T + b1)        [8192, 4096] bf16
  gemm_bf16_nt<true, true, false><<<dim3(DFF / 128, M / 128), 256, 0,
                                    stream>>>(x1_b, w1_b, b1, nullptr, h1_b, M,
                                              DFF, E);

  // ff = h1 @ w2^T + b2              [8192, 1024] f32
  gemm_bf16_nt<false, false, true><<<dim3(E / 128, M / 128), 256, 0, stream>>>(
      h1_b, w2_b, b2, ff_f, nullptr, M, E, DFF);

  // out = LN(x1 + ff)
  add_ln<<<M, 256, 0, stream>>>(x1_f, ff_f, ln2_g, ln2_b, x_out, nullptr);
}
